// Net_83210696393125
// MI455X (gfx1250) — matrix operand order repaired and checked by reading, not compiled, not run on hardware
//
#include <hip/hip_runtime.h>
#include <hip/hip_bf16.h>

typedef __attribute__((ext_vector_type(16))) _Float16 v16h;
typedef __attribute__((ext_vector_type(8)))  float    v8f;

#define Bb 1024
#define Tt 30
#define Dd 300
#define Kk 32
#define TXT 9600
#define Ee 512
#define DI 2048
#define DM 1024
#define DV 2048
#define SI 30
#define SV 30

// Async global->LDS staging helpers (CDNA5 GLOBAL_LOAD_ASYNC_TO_LDS, ASYNCcnt)
__device__ __forceinline__ void async_b128(unsigned lds_off, const void* gptr) {
  asm volatile("global_load_async_to_lds_b128 %0, %1, off"
               :: "v"(lds_off), "v"(gptr) : "memory");
}
__device__ __forceinline__ void async_b128_o16(unsigned lds_off, const void* gptr) {
  // INST_OFFSET applies to both LDS and global address (ISA 08_async_tensor §4.4)
  asm volatile("global_load_async_to_lds_b128 %0, %1, off offset:16"
               :: "v"(lds_off), "v"(gptr) : "memory");
}
__device__ __forceinline__ void wait_async0() {
  asm volatile("s_wait_asynccnt 0x0" ::: "memory");
}
__device__ __forceinline__ unsigned lds_off32(const void* p) {
  // generic LDS pointer: low 32 bits are the wave-relative LDS byte address
  return (unsigned)(size_t)p;
}

// ---------------------------------------------------------------------------
// Streaming reductions (memory-bound part: ~500MB of video features)
// ---------------------------------------------------------------------------
__global__ __launch_bounds__(256) void mean_kernel(const float* __restrict__ in,
                                                   _Float16* __restrict__ out,
                                                   int S, int D) {
  size_t i = (size_t)blockIdx.x * 256 + threadIdx.x;
  if (i >= (size_t)Bb * D) return;
  size_t b = i / D, d = i % D;
  float s = 0.f;
  for (int t = 0; t < S; ++t) s += in[((size_t)b * S + t) * D + d];
  out[i] = (_Float16)(s / (float)S);
}

__global__ __launch_bounds__(256) void max_kernel(const float* __restrict__ in,
                                                  _Float16* __restrict__ out,
                                                  int S, int D) {
  size_t i = (size_t)blockIdx.x * 256 + threadIdx.x;
  if (i >= (size_t)Bb * D) return;
  size_t b = i / D, d = i % D;
  float s = -__builtin_inff();
  for (int t = 0; t < S; ++t) s = fmaxf(s, in[((size_t)b * S + t) * D + d]);
  out[i] = (_Float16)s;
}

__global__ __launch_bounds__(256) void cvt_kernel(const float* __restrict__ in,
                                                  _Float16* __restrict__ out, size_t n) {
  size_t i = (size_t)blockIdx.x * 256 + threadIdx.x;
  if (i < n) out[i] = (_Float16)in[i];
}

// W [K,N] f32 -> Wt [N,K] f16  (so WMMA B-fragments read contiguous K)
__global__ __launch_bounds__(256) void transpose_cvt_kernel(const float* __restrict__ W,
                                                            _Float16* __restrict__ Wt,
                                                            int K, int N) {
  size_t i = (size_t)blockIdx.x * 256 + threadIdx.x;
  if (i >= (size_t)K * N) return;
  size_t n = i / K, k = i % K;
  Wt[i] = (_Float16)W[k * (size_t)N + n];
}

// ---------------------------------------------------------------------------
// NetVLAD: one workgroup per batch row. Two-pass vlad (recompute) to keep
// static LDS ~40KB. Writes L2-normalized tfeat directly as f16.
// ---------------------------------------------------------------------------
__global__ __launch_bounds__(256) void netvlad_kernel(const float* __restrict__ text,
                                                      const float* __restrict__ nvc,
                                                      const float* __restrict__ nvc2,
                                                      const float* __restrict__ nvg,
                                                      const float* __restrict__ nvb,
                                                      _Float16* __restrict__ tfeat16) {
  __shared__ float tx[Tt][Dd];
  __shared__ float asg[Tt][Kk];
  __shared__ float asum[Kk];
  __shared__ float colnorm2[Kk];
  __shared__ float colinv[Kk];
  __shared__ float tnorm_inv;
  const int b = blockIdx.x, tid = threadIdx.x;
  const float bnscale = rsqrtf(1.f + 1e-5f);

  for (int i = tid; i < Tt * Dd; i += 256)
    tx[i / Dd][i % Dd] = text[(size_t)b * Tt * Dd + i];
  if (tid < Kk) colnorm2[tid] = 0.f;
  __syncthreads();

  for (int idx = tid; idx < Tt * Kk; idx += 256) {
    int t = idx >> 5, k = idx & 31;
    float s = 0.f;
    for (int d = 0; d < Dd; ++d) s = fmaf(tx[t][d], nvc[d * Kk + k], s);
    asg[t][k] = s * (nvg[k] * bnscale) + nvb[k];
  }
  __syncthreads();
  if (tid < Tt) {
    float m = -__builtin_inff();
    for (int k = 0; k < Kk; ++k) m = fmaxf(m, asg[tid][k]);
    float s = 0.f;
    for (int k = 0; k < Kk; ++k) { float e = __expf(asg[tid][k] - m); asg[tid][k] = e; s += e; }
    float inv = 1.f / s;
    for (int k = 0; k < Kk; ++k) asg[tid][k] *= inv;
  }
  __syncthreads();
  if (tid < Kk) {
    float s = 0.f;
    for (int t = 0; t < Tt; ++t) s += asg[t][tid];
    asum[tid] = s;
  }
  __syncthreads();
  for (int idx = tid; idx < Dd * Kk; idx += 256) {
    int d = idx >> 5, k = idx & 31;
    float v = -asum[k] * nvc2[idx];
    for (int t = 0; t < Tt; ++t) v = fmaf(asg[t][k], tx[t][d], v);
    atomicAdd(&colnorm2[k], v * v);
  }
  __syncthreads();
  if (tid < Kk) colinv[tid] = 1.f / fmaxf(sqrtf(colnorm2[tid]), 1e-12f);
  __syncthreads();
  if (tid == 0) {
    float s = 0.f;
    for (int k = 0; k < Kk; ++k) s += colnorm2[k] * colinv[k] * colinv[k];
    tnorm_inv = 1.f / fmaxf(sqrtf(s), 1e-12f);
  }
  __syncthreads();
  for (int idx = tid; idx < Dd * Kk; idx += 256) {
    int d = idx >> 5, k = idx & 31;
    float v = -asum[k] * nvc2[idx];
    for (int t = 0; t < Tt; ++t) v = fmaf(asg[t][k], tx[t][d], v);
    tfeat16[(size_t)b * TXT + idx] = (_Float16)(v * colinv[k] * tnorm_inv);
  }
}

// ---------------------------------------------------------------------------
// MoE softmax weights: [B,3] = softmax(tfeat @ moeW + moeb)
// ---------------------------------------------------------------------------
__global__ __launch_bounds__(256) void moe_kernel(const _Float16* __restrict__ tfeat16,
                                                  const float* __restrict__ moeW,
                                                  const float* __restrict__ moeb,
                                                  float* __restrict__ moe) {
  __shared__ float red[3][256];
  const int b = blockIdx.x, tid = threadIdx.x;
  float s0 = 0.f, s1 = 0.f, s2 = 0.f;
  for (int i = tid; i < TXT; i += 256) {
    float t = (float)tfeat16[(size_t)b * TXT + i];
    s0 = fmaf(t, moeW[i * 3 + 0], s0);
    s1 = fmaf(t, moeW[i * 3 + 1], s1);
    s2 = fmaf(t, moeW[i * 3 + 2], s2);
  }
  red[0][tid] = s0; red[1][tid] = s1; red[2][tid] = s2;
  __syncthreads();
  for (int off = 128; off; off >>= 1) {
    if (tid < off) {
      red[0][tid] += red[0][tid + off];
      red[1][tid] += red[1][tid + off];
      red[2][tid] += red[2][tid + off];
    }
    __syncthreads();
  }
  if (tid == 0) {
    float l0 = red[0][0] + moeb[0], l1 = red[1][0] + moeb[1], l2 = red[2][0] + moeb[2];
    float m = fmaxf(l0, fmaxf(l1, l2));
    float e0 = __expf(l0 - m), e1 = __expf(l1 - m), e2 = __expf(l2 - m);
    float inv = 1.f / (e0 + e1 + e2);
    moe[b * 3 + 0] = e0 * inv; moe[b * 3 + 1] = e1 * inv; moe[b * 3 + 2] = e2 * inv;
  }
}

// ---------------------------------------------------------------------------
// WMMA GEMM: C[M,N]f32 = A[M,K]f16 @ Wt[N,K]f16 (+bias), optional f16 copy.
// 8 waves (2x4), wave tile 32x32 (2x2 WMMA subtiles), block tile 64x128.
// Double-buffered LDS, async global->LDS staging, one barrier per k-step.
// ---------------------------------------------------------------------------
__global__ __launch_bounds__(256) void gemm_f16_kernel(const _Float16* __restrict__ A,
                                                       const _Float16* __restrict__ Wt,
                                                       const float* __restrict__ bias,
                                                       float* __restrict__ C,
                                                       _Float16* __restrict__ Ch,
                                                       int M, int N, int K) {
  __shared__ __align__(16) _Float16 As[2][64][32];    // 8KB
  __shared__ __align__(16) _Float16 Bs[2][128][32];   // 16KB
  const int tid = threadIdx.x, lane = tid & 31, wave = tid >> 5;
  const int wm = wave >> 2, wn = wave & 3;
  const int m0 = blockIdx.y * 64, n0 = blockIdx.x * 128;
  // staging coords: A 64x32 halves -> 1 x b128/thread; B 128x32 -> 2 x b128/thread
  const int ar = tid >> 2, ac = (tid & 3) * 8;
  const int br = tid >> 1, bc = (tid & 1) * 16;

  v8f acc[2][2] = {{{}, {}}, {{}, {}}};

  auto stage = [&](int buf, int k0) {
    async_b128(lds_off32(&As[buf][ar][ac]), A + (size_t)(m0 + ar) * K + k0 + ac);
    unsigned lb = lds_off32(&Bs[buf][br][bc]);
    const _Float16* gb = Wt + (size_t)(n0 + br) * K + k0 + bc;
    async_b128(lb, gb);
    async_b128_o16(lb, gb);
  };

  stage(0, 0);
  wait_async0();
  __syncthreads();

  const int nk = K >> 5;
  const int row0 = wm * 32 + (lane & 15);
  const int col0 = wn * 32 + (lane & 15);
  const int half = lane >> 4;

  for (int ks = 0; ks < nk; ++ks) {
    const int cur = ks & 1, nxt = cur ^ 1;
    if (ks + 1 < nk) stage(nxt, (ks + 1) * 32);

    v16h a0, a1, b0, b1;
#pragma unroll
    for (int v = 0; v < 8; ++v) {
      int kk = (v < 4) ? (2 * v + 8 * half) : (16 + 2 * (v - 4) + 8 * half);
      a0[2 * v] = As[cur][row0][kk];      a0[2 * v + 1] = As[cur][row0][kk + 1];
      a1[2 * v] = As[cur][row0 + 16][kk]; a1[2 * v + 1] = As[cur][row0 + 16][kk + 1];
    }
#pragma unroll
    for (int e = 0; e < 16; ++e) {
      b0[e] = Bs[cur][col0][8 * half + e + (e & 8)];
      b1[e] = Bs[cur][col0 + 16][8 * half + e + (e & 8)];
    }
    acc[0][0] = __builtin_amdgcn_wmma_f32_16x16x32_f16(false, a0, false, b0, (short)0, acc[0][0], false, false);
    acc[0][1] = __builtin_amdgcn_wmma_f32_16x16x32_f16(false, a0, false, b1, (short)0, acc[0][1], false, false);
    acc[1][0] = __builtin_amdgcn_wmma_f32_16x16x32_f16(false, a1, false, b0, (short)0, acc[1][0], false, false);
    acc[1][1] = __builtin_amdgcn_wmma_f32_16x16x32_f16(false, a1, false, b1, (short)0, acc[1][1], false, false);

    wait_async0();
    __syncthreads();
  }

#pragma unroll
  for (int sn = 0; sn < 2; ++sn) {
    int col = n0 + wn * 32 + sn * 16 + (lane & 15);
    float bv = bias ? bias[col] : 0.f;
#pragma unroll
    for (int sm = 0; sm < 2; ++sm) {
#pragma unroll
      for (int r = 0; r < 8; ++r) {
        int row = m0 + wm * 32 + sm * 16 + r + 8 * half;  // C layout: M = r + 8*half
        float v = acc[sm][sn][r] + bv;
        C[(size_t)row * N + col] = v;
        if (Ch) Ch[(size_t)row * N + col] = (_Float16)v;
      }
    }
  }
}

// ---------------------------------------------------------------------------
// Context gating epilogue: emb = l2norm(x * sigmoid(bn(g1))), f16 out.
// ---------------------------------------------------------------------------
__global__ __launch_bounds__(256) void gate_kernel(const float* __restrict__ x,
                                                   const float* __restrict__ g1,
                                                   const float* __restrict__ g,
                                                   const float* __restrict__ bb,
                                                   _Float16* __restrict__ emb16) {
  __shared__ float red[256];
  const int b = blockIdx.x, tid = threadIdx.x;
  const float bnscale = rsqrtf(1.f + 1e-5f);
  float y[2];
  float nrm = 0.f;
#pragma unroll
  for (int j = 0; j < 2; ++j) {
    int n = tid + j * 256;
    float xv = x[(size_t)b * Ee + n];
    float gv = g1[(size_t)b * Ee + n] * (g[n] * bnscale) + bb[n];
    float yv = xv / (1.f + __expf(-gv));
    y[j] = yv; nrm += yv * yv;
  }
  red[tid] = nrm;
  __syncthreads();
  for (int off = 128; off; off >>= 1) {
    if (tid < off) red[tid] += red[tid + off];
    __syncthreads();
  }
  float inv = 1.f / fmaxf(sqrtf(red[0]), 1e-12f);
#pragma unroll
  for (int j = 0; j < 2; ++j)
    emb16[(size_t)b * Ee + tid + j * 256] = (_Float16)(y[j] * inv);
}

// ---------------------------------------------------------------------------
// Fused conf: three WMMA Gram matrices (K=512) + MoE/availability epilogue.
// Block tile 64x128, wave tile 32x32 per modality (12 WMMAs per k-step).
// Single-buffered async staging (36KB LDS).
// ---------------------------------------------------------------------------
__global__ __launch_bounds__(256) void conf_kernel(const _Float16* __restrict__ t0,
                                                   const _Float16* __restrict__ t1,
                                                   const _Float16* __restrict__ t2,
                                                   const _Float16* __restrict__ v0,
                                                   const _Float16* __restrict__ v1,
                                                   const _Float16* __restrict__ v2,
                                                   const float* __restrict__ moe,
                                                   const int* __restrict__ indI,
                                                   const int* __restrict__ indM,
                                                   const int* __restrict__ indV,
                                                   float* __restrict__ out) {
  __shared__ __align__(16) _Float16 As[3][64][32];    // 12KB
  __shared__ __align__(16) _Float16 Bs[3][128][32];   // 24KB
  const int tid = threadIdx.x, lane = tid & 31, wave = tid >> 5;
  const int wm = wave >> 2, wn = wave & 3;
  const int m0 = blockIdx.y * 64, n0 = blockIdx.x * 128;
  const _Float16* Ts[3] = {t0, t1, t2};
  const _Float16* Vs[3] = {v0, v1, v2};
  const int ar = tid >> 2, ac = (tid & 3) * 8;
  const int br = tid >> 1, bc = (tid & 1) * 16;
  const int row0 = wm * 32 + (lane & 15);
  const int col0 = wn * 32 + (lane & 15);
  const int half = lane >> 4;

  v8f acc[3][2][2] = {};

  for (int k0 = 0; k0 < Ee; k0 += 32) {
#pragma unroll
    for (int j = 0; j < 3; ++j) {
      async_b128(lds_off32(&As[j][ar][ac]), Ts[j] + (size_t)(m0 + ar) * Ee + k0 + ac);
      unsigned lb = lds_off32(&Bs[j][br][bc]);
      const _Float16* gb = Vs[j] + (size_t)(n0 + br) * Ee + k0 + bc;
      async_b128(lb, gb);
      async_b128_o16(lb, gb);
    }
    wait_async0();
    __syncthreads();
#pragma unroll
    for (int j = 0; j < 3; ++j) {
      v16h a0, a1, b0, b1;
#pragma unroll
      for (int v = 0; v < 8; ++v) {
        int kk = (v < 4) ? (2 * v + 8 * half) : (16 + 2 * (v - 4) + 8 * half);
        a0[2 * v] = As[j][row0][kk];      a0[2 * v + 1] = As[j][row0][kk + 1];
        a1[2 * v] = As[j][row0 + 16][kk]; a1[2 * v + 1] = As[j][row0 + 16][kk + 1];
      }
#pragma unroll
      for (int e = 0; e < 16; ++e) {
        b0[e] = Bs[j][col0][8 * half + e + (e & 8)];
        b1[e] = Bs[j][col0 + 16][8 * half + e + (e & 8)];
      }
      acc[j][0][0] = __builtin_amdgcn_wmma_f32_16x16x32_f16(false, a0, false, b0, (short)0, acc[j][0][0], false, false);
      acc[j][0][1] = __builtin_amdgcn_wmma_f32_16x16x32_f16(false, a0, false, b1, (short)0, acc[j][0][1], false, false);
      acc[j][1][0] = __builtin_amdgcn_wmma_f32_16x16x32_f16(false, a1, false, b0, (short)0, acc[j][1][0], false, false);
      acc[j][1][1] = __builtin_amdgcn_wmma_f32_16x16x32_f16(false, a1, false, b1, (short)0, acc[j][1][1], false, false);
    }
    __syncthreads();
  }

#pragma unroll
  for (int sn = 0; sn < 2; ++sn) {
    int col = n0 + wn * 32 + sn * 16 + (lane & 15);
    float a0 = indI[col] ? 1.f : 0.f;
    float a1 = indM[col] ? 1.f : 0.f;
    float a2 = indV[col] ? 1.f : 0.f;
#pragma unroll
    for (int sm = 0; sm < 2; ++sm) {
#pragma unroll
      for (int r = 0; r < 8; ++r) {
        int row = m0 + wm * 32 + sm * 16 + r + 8 * half;
        float w0 = a0 * moe[row * 3 + 0];
        float w1 = a1 * moe[row * 3 + 1];
        float w2 = a2 * moe[row * 3 + 2];
        float den = w0 + w1 + w2;
        out[(size_t)row * Bb + col] =
            (w0 * acc[0][sm][sn][r] + w1 * acc[1][sm][sn][r] + w2 * acc[2][sm][sn][r]) / den;
      }
    }
  }
}

// ---------------------------------------------------------------------------
// Host launcher
// ---------------------------------------------------------------------------
extern "C" void kernel_launch(void* const* d_in, const int* in_sizes, int n_in,
                              void* d_out, int out_size, void* d_ws, size_t ws_size,
                              hipStream_t stream) {
  const float* text    = (const float*)d_in[0];
  const float* vinst   = (const float*)d_in[1];
  const float* vmot    = (const float*)d_in[2];
  const float* vvis    = (const float*)d_in[3];
  const float* nvc     = (const float*)d_in[4];
  const float* nvc2    = (const float*)d_in[5];
  const float* nvg     = (const float*)d_in[6];
  const float* nvb     = (const float*)d_in[7];
  const float* moeW    = (const float*)d_in[8];
  const float* moeb    = (const float*)d_in[9];
  const float* tW      = (const float*)d_in[10];
  const float* tb      = (const float*)d_in[11];
  const float* tcW     = (const float*)d_in[12];
  const float* tcb     = (const float*)d_in[13];
  const float* tg      = (const float*)d_in[14];
  const float* tbb     = (const float*)d_in[15];
  const float* iW      = (const float*)d_in[16];
  const float* ib      = (const float*)d_in[17];
  const float* icW     = (const float*)d_in[18];
  const float* icb     = (const float*)d_in[19];
  const float* ig      = (const float*)d_in[20];
  const float* ibb     = (const float*)d_in[21];
  const float* mW      = (const float*)d_in[22];
  const float* mb      = (const float*)d_in[23];
  const float* mcW     = (const float*)d_in[24];
  const float* mcb     = (const float*)d_in[25];
  const float* mg      = (const float*)d_in[26];
  const float* mbb     = (const float*)d_in[27];
  const float* vW      = (const float*)d_in[28];
  const float* vb      = (const float*)d_in[29];
  const float* vcW     = (const float*)d_in[30];
  const float* vcb     = (const float*)d_in[31];
  const float* vg      = (const float*)d_in[32];
  const float* vbb     = (const float*)d_in[33];
  const int*   indI    = (const int*)d_in[34];
  const int*   indM    = (const int*)d_in[35];
  const int*   indV    = (const int*)d_in[36];
  float* out = (float*)d_out;

  char* ws = (char*)d_ws;
  size_t off = 0;
  auto alloc = [&](size_t bytes) -> void* {
    void* p = ws + off;
    off = (off + bytes + 255) & ~(size_t)255;
    return p;
  };
  _Float16* tfeat16 = (_Float16*)alloc((size_t)Bb * TXT * 2);
  _Float16* aggI16  = (_Float16*)alloc((size_t)Bb * DI * 2);
  _Float16* aggM16  = (_Float16*)alloc((size_t)Bb * DM * 2);
  _Float16* aggV16  = (_Float16*)alloc((size_t)Bb * DV * 2);
  _Float16* tWt16   = (_Float16*)alloc((size_t)3 * Ee * TXT * 2);
  _Float16* iWt16   = (_Float16*)alloc((size_t)Ee * DI * 2);
  _Float16* mWt16   = (_Float16*)alloc((size_t)Ee * DM * 2);
  _Float16* vWt16   = (_Float16*)alloc((size_t)Ee * DV * 2);
  _Float16* tcWt16  = (_Float16*)alloc((size_t)3 * Ee * Ee * 2);
  _Float16* icWt16  = (_Float16*)alloc((size_t)Ee * Ee * 2);
  _Float16* mcWt16  = (_Float16*)alloc((size_t)Ee * Ee * 2);
  _Float16* vcWt16  = (_Float16*)alloc((size_t)Ee * Ee * 2);
  float*    moe     = (float*)alloc((size_t)Bb * 3 * 4);
  float*    bufX    = (float*)alloc((size_t)Bb * Ee * 4);
  _Float16* bufX16  = (_Float16*)alloc((size_t)Bb * Ee * 2);
  float*    bufG    = (float*)alloc((size_t)Bb * Ee * 4);
  _Float16* embVI   = (_Float16*)alloc((size_t)Bb * Ee * 2);
  _Float16* embVM   = (_Float16*)alloc((size_t)Bb * Ee * 2);
  _Float16* embVV   = (_Float16*)alloc((size_t)Bb * Ee * 2);
  _Float16* embT0   = (_Float16*)alloc((size_t)Bb * Ee * 2);
  _Float16* embT1   = (_Float16*)alloc((size_t)Bb * Ee * 2);
  _Float16* embT2   = (_Float16*)alloc((size_t)Bb * Ee * 2);
  (void)ws_size; (void)n_in; (void)in_sizes; (void)out_size;

  // ---- 1. pooling / conversions ----
  {
    int n = Bb * DI;
    mean_kernel<<<(n + 255) / 256, 256, 0, stream>>>(vinst, aggI16, SI, DI);
  }
  {
    int n = Bb * DV;
    max_kernel<<<(n + 255) / 256, 256, 0, stream>>>(vvis, aggV16, SV, DV);
  }
  {
    size_t n = (size_t)Bb * DM;
    cvt_kernel<<<(unsigned)((n + 255) / 256), 256, 0, stream>>>(vmot, aggM16, n);
  }
  // ---- 2. weight transpose+convert ----
  for (int j = 0; j < 3; ++j) {
    size_t n = (size_t)TXT * Ee;
    transpose_cvt_kernel<<<(unsigned)((n + 255) / 256), 256, 0, stream>>>(
        tW + (size_t)j * TXT * Ee, tWt16 + (size_t)j * Ee * TXT, TXT, Ee);
  }
  {
    size_t n = (size_t)DI * Ee;
    transpose_cvt_kernel<<<(unsigned)((n + 255) / 256), 256, 0, stream>>>(iW, iWt16, DI, Ee);
  }
  {
    size_t n = (size_t)DM * Ee;
    transpose_cvt_kernel<<<(unsigned)((n + 255) / 256), 256, 0, stream>>>(mW, mWt16, DM, Ee);
  }
  {
    size_t n = (size_t)DV * Ee;
    transpose_cvt_kernel<<<(unsigned)((n + 255) / 256), 256, 0, stream>>>(vW, vWt16, DV, Ee);
  }
  {
    size_t n = (size_t)Ee * Ee;
    for (int j = 0; j < 3; ++j)
      transpose_cvt_kernel<<<(unsigned)((n + 255) / 256), 256, 0, stream>>>(
          tcW + (size_t)j * Ee * Ee, tcWt16 + (size_t)j * Ee * Ee, Ee, Ee);
    transpose_cvt_kernel<<<(unsigned)((n + 255) / 256), 256, 0, stream>>>(icW, icWt16, Ee, Ee);
    transpose_cvt_kernel<<<(unsigned)((n + 255) / 256), 256, 0, stream>>>(mcW, mcWt16, Ee, Ee);
    transpose_cvt_kernel<<<(unsigned)((n + 255) / 256), 256, 0, stream>>>(vcW, vcWt16, Ee, Ee);
  }
  // ---- 3. NetVLAD + MoE ----
  netvlad_kernel<<<Bb, 256, 0, stream>>>(text, nvc, nvc2, nvg, nvb, tfeat16);
  moe_kernel<<<Bb, 256, 0, stream>>>(tfeat16, moeW, moeb, moe);

  // ---- 4. Gated Embedding Units (WMMA GEMMs + gating epilogue) ----
  const dim3 gE(Ee / 128, Bb / 64);   // N=512 -> 4, M=1024 -> 16
  // instance
  gemm_f16_kernel<<<gE, 256, 0, stream>>>(aggI16, iWt16, ib, bufX, bufX16, Bb, Ee, DI);
  gemm_f16_kernel<<<gE, 256, 0, stream>>>(bufX16, icWt16, icb, bufG, (_Float16*)nullptr, Bb, Ee, Ee);
  gate_kernel<<<Bb, 256, 0, stream>>>(bufX, bufG, ig, ibb, embVI);
  // motion
  gemm_f16_kernel<<<gE, 256, 0, stream>>>(aggM16, mWt16, mb, bufX, bufX16, Bb, Ee, DM);
  gemm_f16_kernel<<<gE, 256, 0, stream>>>(bufX16, mcWt16, mcb, bufG, (_Float16*)nullptr, Bb, Ee, Ee);
  gate_kernel<<<Bb, 256, 0, stream>>>(bufX, bufG, mg, mbb, embVM);
  // visual
  gemm_f16_kernel<<<gE, 256, 0, stream>>>(aggV16, vWt16, vb, bufX, bufX16, Bb, Ee, DV);
  gemm_f16_kernel<<<gE, 256, 0, stream>>>(bufX16, vcWt16, vcb, bufG, (_Float16*)nullptr, Bb, Ee, Ee);
  gate_kernel<<<Bb, 256, 0, stream>>>(bufX, bufG, vg, vbb, embVV);
  // text branches
  _Float16* embT[3] = {embT0, embT1, embT2};
  for (int j = 0; j < 3; ++j) {
    gemm_f16_kernel<<<gE, 256, 0, stream>>>(tfeat16, tWt16 + (size_t)j * Ee * TXT,
                                            tb + (size_t)j * Ee, bufX, bufX16, Bb, Ee, TXT);
    gemm_f16_kernel<<<gE, 256, 0, stream>>>(bufX16, tcWt16 + (size_t)j * Ee * Ee,
                                            tcb + (size_t)j * Ee, bufG, (_Float16*)nullptr,
                                            Bb, Ee, Ee);
    gate_kernel<<<Bb, 256, 0, stream>>>(bufX, bufG, tg + (size_t)j * Ee,
                                        tbb + (size_t)j * Ee, embT[j]);
  }

  // ---- 5. fused conf ----
  const dim3 gC(Bb / 128, Bb / 64);   // N=1024 -> 8, M=1024 -> 16
  conf_kernel<<<gC, 256, 0, stream>>>(embT0, embT1, embT2, embVI, embVM, embVV,
                                      moe, indI, indM, indV, out);
}
